// PhaseCoherenceComputer_90632399880438
// MI455X (gfx1250) — matrix operand order repaired and checked by reading, not compiled, not run on hardware
//
#include <hip/hip_runtime.h>

// PhaseCoherenceComputer for MI455X (gfx1250, wave32, WMMA).
//   D[bh] = (cos(Q)cos(K)^T + sin(Q)sin(K)^T)/64  ==  F(Q) F(K)^T / 64,
//   F(x) = [cos(x) | sin(x)]  (K-dim = 128), batched GEMM M=N=2048, batch=16.
// Output (256 MiB f32) dominates: HBM-bound at ~11.5us @ 23.3 TB/s.
// f16 WMMA compute (~17.2 GFLOP) is ~3x cheaper than the store -> simple
// L2-fed v_wmma_f32_16x16x32_f16 tiling is the right design point.

#define NUM_H 64
#define KDIM  128      // 2*NUM_H: [cos | sin]
#define SEQ   2048
#define BH    16       // 2 batches * 8 heads

typedef _Float16 half16 __attribute__((ext_vector_type(16)));
typedef _Float16 half8  __attribute__((ext_vector_type(8)));
typedef float    float8 __attribute__((ext_vector_type(8)));

// ---------------------------------------------------------------------------
// Phase 1: f32 phases [BH][SEQ][64] -> f16 features [BH][SEQ][128] = cos|sin
// ---------------------------------------------------------------------------
__global__ __launch_bounds__(256) void phase_features_f16(
    const float* __restrict__ ph, _Float16* __restrict__ out) {
  int idx = blockIdx.x * 256 + threadIdx.x;   // [0, BH*SEQ*64)
  int d   = idx & (NUM_H - 1);
  int row = idx >> 6;                         // bh*SEQ + s
  float s, c;
  __sincosf(ph[idx], &s, &c);
  _Float16* o = out + (size_t)row * KDIM;
  o[d]         = (_Float16)c;
  o[NUM_H + d] = (_Float16)s;
}

// ---------------------------------------------------------------------------
// Phase 2: batched GEMM D = Qf * Kf^T / 64 via v_wmma_f32_16x16x32_f16.
// One wave -> one 32x32 output tile (2x2 accumulators, 4 K-chunks of 32).
// ---------------------------------------------------------------------------
__global__ __launch_bounds__(256) void coherence_wmma(
    const _Float16* __restrict__ Q, const _Float16* __restrict__ K,
    float* __restrict__ D) {
  const int lane = threadIdx.x & 31;
  const int wave = threadIdx.x >> 5;
  const int tile = blockIdx.x * 8 + wave;     // 65536 wave-tiles total
  const int bh   = tile >> 12;                // / (64*64 tiles per head)
  const int rem  = tile & 4095;
  const int m0   = (rem >> 6) * 32;
  const int n0   = (rem & 63) * 32;

  const int l = lane & 15;                    // row-in-tile (A) / col (B,D)
  const int g = lane >> 4;                    // K/M half selector

  const _Float16* qb = Q + (size_t)bh * SEQ * KDIM;
  const _Float16* kb = K + (size_t)bh * SEQ * KDIM;

  float8 c00 = {}, c01 = {}, c10 = {}, c11 = {};

#pragma unroll
  for (int kc = 0; kc < 4; ++kc) {
    const int k0 = kc * 32;
    // A fragments: lane l holds row (m0 + 16*mg + l); lanes 0-15 carry K
    // k0..k0+15, lanes 16-31 carry k0+16..k0+31 (contiguous).
    const _Float16* ar0 = qb + (size_t)(m0 + l) * KDIM + k0 + 16 * g;
    const _Float16* ar1 = ar0 + (size_t)16 * KDIM;
    half8 a0lo = *(const half8*)(ar0);
    half8 a0hi = *(const half8*)(ar0 + 8);
    half8 a1lo = *(const half8*)(ar1);
    half8 a1hi = *(const half8*)(ar1 + 8);
    half16 a0 = __builtin_shufflevector(a0lo, a0hi,
        0,1,2,3,4,5,6,7,8,9,10,11,12,13,14,15);
    half16 a1 = __builtin_shufflevector(a1lo, a1hi,
        0,1,2,3,4,5,6,7,8,9,10,11,12,13,14,15);
    // B fragments (K^T): lane (g,l) holds 16 contiguous K-features of row
    // n0 + 16*ng + l starting at k0 + 16*g  -> one aligned 32B load each.
    half16 b0 = *(const half16*)(kb + (size_t)(n0 + l) * KDIM + k0 + 16 * g);
    half16 b1 = *(const half16*)(kb + (size_t)(n0 + 16 + l) * KDIM + k0 + 16 * g);

    c00 = __builtin_amdgcn_wmma_f32_16x16x32_f16(false, a0, false, b0,
                                                 (short)0, c00, false, false);
    c01 = __builtin_amdgcn_wmma_f32_16x16x32_f16(false, a0, false, b1,
                                                 (short)0, c01, false, false);
    c10 = __builtin_amdgcn_wmma_f32_16x16x32_f16(false, a1, false, b0,
                                                 (short)0, c10, false, false);
    c11 = __builtin_amdgcn_wmma_f32_16x16x32_f16(false, a1, false, b1,
                                                 (short)0, c11, false, false);
  }

  // C/D layout: VGPR r <-> row m0 + 8*g + r (+16 per mg), col n0 + l (+16 per ng)
  const float scale = 1.0f / (float)NUM_H;    // TEMPERATURE == 1.0
  float* dOut = D + (size_t)bh * SEQ * SEQ;
#pragma unroll
  for (int r = 0; r < 8; ++r) {
    const size_t row0 = (size_t)(m0 + 8 * g + r) * SEQ;
    const size_t row1 = row0 + (size_t)16 * SEQ;
    const int col0 = n0 + l;
    dOut[row0 + col0]      = c00[r] * scale;
    dOut[row0 + col0 + 16] = c01[r] * scale;
    dOut[row1 + col0]      = c10[r] * scale;
    dOut[row1 + col0 + 16] = c11[r] * scale;
  }
}

// ---------------------------------------------------------------------------
extern "C" void kernel_launch(void* const* d_in, const int* in_sizes, int n_in,
                              void* d_out, int out_size, void* d_ws,
                              size_t ws_size, hipStream_t stream) {
  const float* pq = (const float*)d_in[0];    // phases_q [2,8,2048,64] f32
  const float* pk = (const float*)d_in[1];    // phases_k [2,8,2048,64] f32

  _Float16* qh = (_Float16*)d_ws;                       // 8 MiB
  _Float16* kh = qh + (size_t)BH * SEQ * KDIM;          // 8 MiB

  const int nconv = BH * SEQ * NUM_H;                   // 2,097,152
  phase_features_f16<<<nconv / 256, 256, 0, stream>>>(pq, qh);
  phase_features_f16<<<nconv / 256, 256, 0, stream>>>(pk, kh);

  const int nWaveTiles = BH * (SEQ / 32) * (SEQ / 32);  // 65536
  coherence_wmma<<<nWaveTiles / 8, 256, 0, stream>>>(qh, kh, (float*)d_out);
}
